// LitToClauseLayer_13597866459547
// MI455X (gfx1250) — matrix operand order repaired and checked by reading, not compiled, not run on hardware
//
#include <hip/hip_runtime.h>
#include <hip/hip_bf16.h>
#include <math.h>

// ---- CDNA5 (gfx1250) wave32 WMMA / TDM types ----
typedef __attribute__((ext_vector_type(16))) __bf16        v16bf;
typedef __attribute__((ext_vector_type(8)))  __bf16        v8bf;
typedef __attribute__((ext_vector_type(8)))  float         v8f;
typedef __attribute__((ext_vector_type(4)))  unsigned int  v4u;
typedef __attribute__((ext_vector_type(8)))  int           v8i;
typedef __attribute__((ext_vector_type(4)))  int           v4i;

#define D_FEAT    128
#define KTOT      256          // [msg | h0] concatenated along K
#define NGATES    512          // 4 * D
#define MTILE     32           // rows per block (two 16-row WMMA subtiles)
#define LDS_PITCH 264          // 256 bf16 + 8 bf16 (16B) TDM padding -> 528B rows

// -----------------------------------------------------------------------------
// Kernel 1: segment-sum of literal features into clauses (edge_clause sorted),
// fused with bf16 conversion and packing of A = [msg | h0] per row.
// One block (128 threads) per clause; the binary search is block-uniform.
// -----------------------------------------------------------------------------
__global__ void seg_pack_kernel(const float* __restrict__ x_l,
                                const float* __restrict__ h0,
                                const int* __restrict__ edge_lit,
                                const int* __restrict__ edge_clause,
                                __bf16* __restrict__ A,
                                int n_edges) {
  const int c = blockIdx.x;
  const int d = threadIdx.x;  // 0..127

  int lo = 0, hi = n_edges;
  while (lo < hi) {
    int mid = (lo + hi) >> 1;
    if (edge_clause[mid] < c) lo = mid + 1; else hi = mid;
  }
  const int start = lo;
  int lo2 = start, hi2 = n_edges;
  while (lo2 < hi2) {
    int mid = (lo2 + hi2) >> 1;
    if (edge_clause[mid] < c + 1) lo2 = mid + 1; else hi2 = mid;
  }
  const int end = lo2;

  float acc = 0.0f;
  for (int e = start; e < end; ++e) {
    const int lit = edge_lit[e];                  // uniform -> scalar load
    acc += x_l[(size_t)lit * D_FEAT + d];         // coalesced 512B row read
  }

  __bf16* row = A + (size_t)c * KTOT;
  row[d]          = (__bf16)acc;
  row[D_FEAT + d] = (__bf16)h0[(size_t)c * D_FEAT + d];
}

// -----------------------------------------------------------------------------
// Kernel 2: pack Wcat[n] = [W_ih[n] | W_hh[n]] as bf16 (256KB, L2-resident),
// fold the two biases.
// -----------------------------------------------------------------------------
__global__ void wpack_kernel(const float* __restrict__ W_ih,
                             const float* __restrict__ W_hh,
                             const float* __restrict__ b_ih,
                             const float* __restrict__ b_hh,
                             __bf16* __restrict__ Wcat,
                             float* __restrict__ bias_sum) {
  const int idx = blockIdx.x * blockDim.x + threadIdx.x;
  if (idx < NGATES * KTOT) {
    const int n = idx >> 8;
    const int k = idx & (KTOT - 1);
    const float v = (k < D_FEAT) ? W_ih[(size_t)n * D_FEAT + k]
                                 : W_hh[(size_t)n * D_FEAT + (k - D_FEAT)];
    Wcat[idx] = (__bf16)v;
  }
  if (idx < NGATES) bias_sum[idx] = b_ih[idx] + b_hh[idx];
}

// -----------------------------------------------------------------------------
// Kernel 3: fused GEMM (gates = A @ Wcat^T + bias) + LSTM cell epilogue.
// Block = 256 threads = 8 wave32s, one 32-row M-tile per block.
//  - A tile (32 x 256 bf16) staged in LDS by the Tensor Data Mover with
//    pad_enable so rows have a 528B pitch (bank-conflict-free ds_load_b128).
//  - Wave w owns gate columns d = w*16 + (lane&15) of all 4 gates for BOTH
//    16-row subtiles -> each B fragment feeds 2 WMMAs; 64 WMMAs per wave.
//  - B fragments double-buffered so global latency overlaps WMMA issue.
// Fragment layouts used here (wave32; A/B K order is contiguous, not the ISA table order):
//   A 16x32 bf16 : lane L holds M=L&15, K = (L>>4)*16 + {0..15}
//   B 32x16 bf16 : lane L holds N=L&15, K = (L>>4)*16 + {0..15}
//   C/D 16x16 f32: VGPR r, lane L -> (M = r + 8*(L>>4), N = L&15)
// -----------------------------------------------------------------------------
__global__ void __launch_bounds__(256)
lstm_gemm_kernel(const __bf16* __restrict__ A,
                 const __bf16* __restrict__ Wcat,
                 const float* __restrict__ bias_sum,
                 const float* __restrict__ c0,
                 float* __restrict__ h_out,
                 float* __restrict__ c_out,
                 int n_clause) {
  __shared__ __bf16 ldsA[MTILE * LDS_PITCH];   // 16,896 B

  const int tile = blockIdx.x;
  const int lane = threadIdx.x & 31;
  const int wave = threadIdx.x >> 5;   // 0..7
  const int half = lane >> 4;          // 0 or 1
  const int l15  = lane & 15;
  const int m0   = tile * MTILE;

  // ---- Stage A tile into LDS via TDM (wave 0 issues; tail zero-filled) ----
#if __has_builtin(__builtin_amdgcn_tensor_load_to_lds)
  if (threadIdx.x < 32) {
    const unsigned lds_off = (unsigned)(unsigned long long)(const void*)&ldsA[0];
    const unsigned long long ga =
        (unsigned long long)(const void*)(A + (size_t)m0 * KTOT);
    const unsigned rem = (unsigned)(n_clause - m0);     // rows left in tensor
    // D# group 0: count=1 | lds_addr | global_addr | type=2
    v4u g0 = { 1u,
               lds_off,
               (unsigned)(ga & 0xFFFFFFFFu),
               ((unsigned)((ga >> 32) & 0x01FFFFFFu)) | 0x80000000u };
    // D# group 1: data_size=2B(1), pad_enable, pad_interval=128 DW(6),
    //             pad_amount=4 DW(3); tensor_dim0=256, tensor_dim1=rem,
    //             tile_dim0=256, tile_dim1=32, dim0_stride=256
    v8i g1 = { (int)((1u << 16) | (1u << 20) | (6u << 22) | (3u << 25)),
               (int)(256u << 16),                       // tensor_dim0[15:0]
               (int)((rem & 0xFFFFu) << 16),            // dim0[31:16]=0 | dim1 lo
               (int)((rem >> 16) | (256u << 16)),       // dim1 hi | tile_dim0
               (int)32,                                 // tile_dim1 | tile_dim2=0
               (int)256,                                // tensor_dim0_stride lo
               0, 0 };
    v4i gz4 = { 0, 0, 0, 0 };
    v8i gz8 = { 0, 0, 0, 0, 0, 0, 0, 0 };
    // 6-arg variant (clang-23 / therock-10.0 headers): groups 2/3 unused for a
    // 2-D tile -> zero-filled; trailing int32x8 zero-filled; cpol = 0.
    __builtin_amdgcn_tensor_load_to_lds(g0, g1, gz4, gz4, gz8, 0);
#if __has_builtin(__builtin_amdgcn_s_wait_tensorcnt)
    __builtin_amdgcn_s_wait_tensorcnt(0);
#else
    asm volatile("s_wait_tensorcnt 0x0" ::: "memory");
#endif
  }
  __syncthreads();
#else
  // Fallback: cooperative copy with the same padded pitch + tail zero-fill.
  for (int idx = threadIdx.x; idx < MTILE * KTOT; idx += 256) {
    const int rr = idx >> 8, cc = idx & (KTOT - 1);
    const int gr = m0 + rr;
    ldsA[rr * LDS_PITCH + cc] =
        (gr < n_clause) ? A[(size_t)gr * KTOT + cc] : (__bf16)0.0f;
  }
  __syncthreads();
#endif

  const __bf16* Arow0 = ldsA + (size_t)l15 * LDS_PITCH;          // subtile 0
  const __bf16* Arow1 = ldsA + (size_t)(16 + l15) * LDS_PITCH;   // subtile 1
  const __bf16* Wp    = Wcat + (size_t)(wave * 16 + l15) * KTOT; // W row n

  v8f acc[2][4] = {};    // [subtile][gate i,f,g,o]

  // Preload B fragments for kk = 0.
  v16bf bcur[4];
#pragma unroll
  for (int g = 0; g < 4; ++g)
    bcur[g] = *(const v16bf*)(Wp + (size_t)g * D_FEAT * KTOT + half * 16);

#pragma unroll
  for (int kk = 0; kk < KTOT / 32; ++kk) {       // 8 K-steps of 32
    const int kbase = kk * 32;

    // A fragments from LDS (bank-conflict-free with 528B pitch).
    v8bf a0lo = *(const v8bf*)(Arow0 + kbase + half * 16);
    v8bf a0hi = *(const v8bf*)(Arow0 + kbase + 8 + half * 16);
    v8bf a1lo = *(const v8bf*)(Arow1 + kbase + half * 16);
    v8bf a1hi = *(const v8bf*)(Arow1 + kbase + 8 + half * 16);
    v16bf a0 = __builtin_shufflevector(a0lo, a0hi, 0, 1, 2, 3, 4, 5, 6, 7,
                                       8, 9, 10, 11, 12, 13, 14, 15);
    v16bf a1 = __builtin_shufflevector(a1lo, a1hi, 0, 1, 2, 3, 4, 5, 6, 7,
                                       8, 9, 10, 11, 12, 13, 14, 15);

    // Prefetch next K-step's B fragments (double buffer).
    v16bf bnext[4];
    if (kk < KTOT / 32 - 1) {
#pragma unroll
      for (int g = 0; g < 4; ++g)
        bnext[g] = *(const v16bf*)(Wp + (size_t)g * D_FEAT * KTOT +
                                   kbase + 32 + half * 16);
    }

#pragma unroll
    for (int g = 0; g < 4; ++g) {
      acc[0][g] = __builtin_amdgcn_wmma_f32_16x16x32_bf16(
          false, a0, false, bcur[g], (short)0, acc[0][g], false, false);
      acc[1][g] = __builtin_amdgcn_wmma_f32_16x16x32_bf16(
          false, a1, false, bcur[g], (short)0, acc[1][g], false, false);
    }

    if (kk < KTOT / 32 - 1) {
#pragma unroll
      for (int g = 0; g < 4; ++g) bcur[g] = bnext[g];
    }
  }

  // ---- fused LSTM-cell epilogue (PyTorch gate order i, f, g, o) ----
  const int d = wave * 16 + l15;                 // feature column 0..127
  const float bi = bias_sum[0 * D_FEAT + d];
  const float bf = bias_sum[1 * D_FEAT + d];
  const float bg = bias_sum[2 * D_FEAT + d];
  const float bo = bias_sum[3 * D_FEAT + d];

#pragma unroll
  for (int s = 0; s < 2; ++s) {
#pragma unroll
    for (int r = 0; r < 8; ++r) {
      const int m = m0 + s * 16 + r + half * 8;  // C/D layout row
      if (m < n_clause) {
        const float iv = 1.0f / (1.0f + __expf(-(acc[s][0][r] + bi)));
        const float fv = 1.0f / (1.0f + __expf(-(acc[s][1][r] + bf)));
        const float gv = tanhf(acc[s][2][r] + bg);
        const float ov = 1.0f / (1.0f + __expf(-(acc[s][3][r] + bo)));
        const float cn = fv * c0[(size_t)m * D_FEAT + d] + iv * gv;
        const float hn = ov * tanhf(cn);
        h_out[(size_t)m * D_FEAT + d] = hn;
        c_out[(size_t)m * D_FEAT + d] = cn;
      }
    }
  }
}

// -----------------------------------------------------------------------------
// Host-side launch. Inputs (setup_inputs order):
//  0 x_l [N_LIT,128]  1 h0 [N_CLAUSE,128]  2 c0 [N_CLAUSE,128]
//  3 W_ih [512,128]   4 W_hh [512,128]     5 b_ih [512]  6 b_hh [512]
//  7 edge_lit [E]     8 edge_clause [E] (sorted)
// Output: h_new then c_new, each [N_CLAUSE,128] f32, concatenated.
// -----------------------------------------------------------------------------
extern "C" void kernel_launch(void* const* d_in, const int* in_sizes, int n_in,
                              void* d_out, int out_size, void* d_ws, size_t ws_size,
                              hipStream_t stream) {
  const float* x_l    = (const float*)d_in[0];
  const float* h0     = (const float*)d_in[1];
  const float* c0     = (const float*)d_in[2];
  const float* W_ih   = (const float*)d_in[3];
  const float* W_hh   = (const float*)d_in[4];
  const float* b_ih   = (const float*)d_in[5];
  const float* b_hh   = (const float*)d_in[6];
  const int* edge_lit    = (const int*)d_in[7];
  const int* edge_clause = (const int*)d_in[8];

  const int n_clause = in_sizes[1] / D_FEAT;
  const int n_edges  = in_sizes[7];

  // Workspace: A_packed bf16 [n_clause,256], Wcat bf16 [512,256], bias f32[512]
  char* ws = (char*)d_ws;
  const size_t szA  = (size_t)n_clause * KTOT * sizeof(unsigned short);
  const size_t offW = (szA + 255) & ~(size_t)255;
  const size_t szW  = (size_t)NGATES * KTOT * sizeof(unsigned short);
  const size_t offB = (offW + szW + 255) & ~(size_t)255;

  __bf16* A        = (__bf16*)(ws);
  __bf16* Wc       = (__bf16*)(ws + offW);
  float*  bias_sum = (float*)(ws + offB);

  seg_pack_kernel<<<n_clause, D_FEAT, 0, stream>>>(x_l, h0, edge_lit,
                                                   edge_clause, A, n_edges);
  wpack_kernel<<<(NGATES * KTOT + 255) / 256, 256, 0, stream>>>(
      W_ih, W_hh, b_ih, b_hh, Wc, bias_sum);

  float* out   = (float*)d_out;
  float* h_new = out;
  float* c_new = out + (size_t)n_clause * D_FEAT;
  const int ntiles = (n_clause + MTILE - 1) / MTILE;
  lstm_gemm_kernel<<<ntiles, 256, 0, stream>>>(A, Wc, bias_sum, c0,
                                               h_new, c_new, n_clause);
}
